// AccMcCnnTesting_71416716198279
// MI455X (gfx1250) — matrix operand order repaired and checked by reading, not compiled, not run on hardware
//
#include <hip/hip_runtime.h>
#include <hip/hip_bf16.h>
#include <math.h>

typedef __attribute__((ext_vector_type(16))) _Float16 v16h;
typedef __attribute__((ext_vector_type(8)))  float    v8f;

#define HF 182
#define WF 438
#define MO (HF*WF)      // 79716 feature pixels
#define D_NUM 32
#define LROW 392        // LDS row stride in halves (384 + pad)

union AF { v16h v; uint32_t u[8]; };

__device__ __forceinline__ v8f wmma16(v16h a, v16h b, v8f c) {
  return __builtin_amdgcn_wmma_f32_16x16x32_f16(false, a, false, b, (short)0, c,
                                                false, false);
}

// A-matrix 16x32 f16 fragment; rowptr already includes the lane's M-row offset.
// Layout (ISA 7.12.2): lanes 0-15 -> K 0..7 / 16..23, lanes 16-31 -> K 8..15 / 24..31.
__device__ __forceinline__ v16h load_a_frag_ptr(const _Float16* rowptr, int kbase, int half) {
  AF f;
  const _Float16* p = rowptr + kbase;
#pragma unroll
  for (int j = 0; j < 8; ++j) {
    int koff = ((j >> 2) << 4) + (half << 3) + ((j & 3) << 1);
    f.u[j] = *reinterpret_cast<const uint32_t*>(p + koff);
  }
  return f.v;
}

__device__ __forceinline__ v16h load_a_frag(const _Float16* base, int lda, int kbase) {
  int lane = threadIdx.x & 31;
  return load_a_frag_ptr(base + (size_t)(lane & 15) * lda, kbase, lane >> 4);
}

// B-matrix 32x16 f16 fragment from weights stored (N, Ktot) row-major
// (i.e. transposed-B => contiguous b32 loads per VGPR).
__device__ __forceinline__ v16h load_b_frag(const _Float16* w, int ldk, int nbase, int kbase) {
  int lane = threadIdx.x & 31;
  const _Float16* p = w + (size_t)(nbase + (lane & 15)) * ldk + kbase + ((lane >> 4) << 3);
  AF f;
#pragma unroll
  for (int j = 0; j < 8; ++j)
    f.u[j] = *reinterpret_cast<const uint32_t*>(p + ((j >> 2) << 4) + ((j & 3) << 1));
  return f.v;
}

// ---------------------------------------------------------------- utility
__global__ void zero_kernel(uint4* p, size_t n4) {
  size_t i = (size_t)blockIdx.x * blockDim.x + threadIdx.x;
  size_t stride = (size_t)gridDim.x * blockDim.x;
  uint4 z = {0u, 0u, 0u, 0u};
  for (; i < n4; i += stride) p[i] = z;
}

// Convert/pad all weights to f16 GEMM layouts.
__global__ void prep_kernel(const float* __restrict__ w2, const float* __restrict__ w3,
                            const float* __restrict__ w4, const float* __restrict__ w5,
                            const float* __restrict__ fw1, const float* __restrict__ fw2,
                            const float* __restrict__ fw3,
                            _Float16* wc2, _Float16* wc3, _Float16* wc4, _Float16* wc5,
                            _Float16* fw1r, _Float16* fw1s, _Float16* fw2h, _Float16* fw3h) {
  const int NC = 112 * 1152;   // conv weight (O=112, K=9*128)
  const int NF1 = 384 * 128;   // fc1 half (N=384, K=128)
  const int NF = 384 * 384;
  int total = 4 * NC + 2 * NF1 + 2 * NF;
  for (int i = blockIdx.x * blockDim.x + threadIdx.x; i < total;
       i += gridDim.x * blockDim.x) {
    if (i < 4 * NC) {
      int layer = i / NC, j = i % NC;
      int o = j / 1152, k = j % 1152;
      int tap = k / 128, ci = k % 128;
      const float* w = layer == 0 ? w2 : layer == 1 ? w3 : layer == 2 ? w4 : w5;
      _Float16* dst = layer == 0 ? wc2 : layer == 1 ? wc3 : layer == 2 ? wc4 : wc5;
      dst[j] = (_Float16)((ci < 112) ? w[(o * 112 + ci) * 9 + tap] : 0.f);
    } else if (i < 4 * NC + 2 * NF1) {
      int r = i - 4 * NC;
      int which = r / NF1, j = r % NF1;
      int nrow = j / 128, c = j % 128;
      _Float16* dst = which == 0 ? fw1r : fw1s;
      dst[j] = (_Float16)((c < 112) ? fw1[nrow * 224 + which * 112 + c] : 0.f);
    } else {
      int r = i - 4 * NC - 2 * NF1;
      int which = r / NF, j = r % NF;
      const float* src = which == 0 ? fw2 : fw3;
      _Float16* dst = which == 0 ? fw2h : fw3h;
      dst[j] = (_Float16)src[j];
    }
  }
}

// -------------------------------------------------- conv1: 1 -> 112, fp32 VALU
__global__ void conv1_kernel(const float* __restrict__ img, const float* __restrict__ w1,
                             const float* __restrict__ b1, _Float16* __restrict__ out,
                             int Ho, int Wo, int Wi) {
  int idx = blockIdx.x * blockDim.x + threadIdx.x;
  int total = Ho * Wo * 112;
  if (idx >= total) return;
  int c = idx % 112;
  int x = (idx / 112) % Wo;
  int y = idx / (112 * Wo);
  float s = b1[c];
  const float* wp = w1 + c * 9;
#pragma unroll
  for (int ky = 0; ky < 3; ++ky)
#pragma unroll
    for (int kx = 0; kx < 3; ++kx)
      s = fmaf(img[(y + ky) * Wi + (x + kx)], wp[ky * 3 + kx], s);
  out[(size_t)(y * Wo + x) * 128 + c] = (_Float16)fmaxf(s, 0.f);
}

// ------------------------------------- conv layers 2-5: implicit GEMM via WMMA
// in : (Hi, Wi, 128) f16, out: (Ho, Wo, 128) f16, w: (112, 9*128) f16
__global__ void __launch_bounds__(128) convN_kernel(
    const _Float16* __restrict__ in, const float* __restrict__ bias,
    const _Float16* __restrict__ w, _Float16* __restrict__ out,
    int Ho, int Wo, int Wi) {
  int lane = threadIdx.x & 31;
  int wid = threadIdx.x >> 5;
  int Mo = Ho * Wo;
  int mtile = blockIdx.x * 4 + wid;
  if (mtile * 16 >= Mo) return;
  int half = lane >> 4;
  int n = lane & 15;
  int p = mtile * 16 + (lane & 15);
  if (p > Mo - 1) p = Mo - 1;
  int y = p / Wo, x = p % Wo;

  v8f acc[7];
#pragma unroll
  for (int t = 0; t < 7; ++t) {
    float bv = bias[t * 16 + n];
#pragma unroll
    for (int v = 0; v < 8; ++v) acc[t][v] = bv;
  }

#pragma unroll 1
  for (int tap = 0; tap < 9; ++tap) {
    int ky = tap / 3, kx = tap % 3;
    const _Float16* rowptr = in + ((size_t)(y + ky) * Wi + (x + kx)) * 128;
#pragma unroll
    for (int kc = 0; kc < 4; ++kc) {
      v16h a = load_a_frag_ptr(rowptr, kc * 32, half);
#pragma unroll
      for (int t = 0; t < 7; ++t) {
        v16h b = load_b_frag(w, 1152, t * 16, tap * 128 + kc * 32);
        acc[t] = wmma16(a, b, acc[t]);
      }
    }
  }

  int mbase = half * 8;
#pragma unroll
  for (int t = 0; t < 7; ++t)
#pragma unroll
    for (int v = 0; v < 8; ++v) {
      int pp = mtile * 16 + mbase + v;
      if (pp < Mo) out[(size_t)pp * 128 + t * 16 + n] = (_Float16)fmaxf(acc[t][v], 0.f);
    }
}

// ---------------- fc1 halves: A = feat @ fw1r^T   /   B = feat @ fw1s^T + fb1
// feat: (MO, 128) f16; w: (384, 128) f16; out: (MO, 384) f16 (no ReLU yet)
__global__ void __launch_bounds__(128) fc1_kernel(
    const _Float16* __restrict__ feat, const _Float16* __restrict__ w,
    const float* __restrict__ bias, int addBias, _Float16* __restrict__ out) {
  int lane = threadIdx.x & 31;
  int wid = threadIdx.x >> 5;
  int mtile = blockIdx.x * 4 + wid;
  if (mtile * 16 >= MO) return;
  int ng = blockIdx.y;                       // 6 groups of 64 output cols
  int half = lane >> 4, n = lane & 15;
  int p = mtile * 16 + (lane & 15);
  if (p > MO - 1) p = MO - 1;
  const _Float16* rowptr = feat + (size_t)p * 128;

  v8f acc[4];
#pragma unroll
  for (int t = 0; t < 4; ++t) {
    float bv = addBias ? bias[ng * 64 + t * 16 + n] : 0.f;
#pragma unroll
    for (int v = 0; v < 8; ++v) acc[t][v] = bv;
  }
#pragma unroll
  for (int kc = 0; kc < 4; ++kc) {
    v16h a = load_a_frag_ptr(rowptr, kc * 32, half);
#pragma unroll
    for (int t = 0; t < 4; ++t) {
      v16h b = load_b_frag(w, 128, ng * 64 + t * 16, kc * 32);
      acc[t] = wmma16(a, b, acc[t]);
    }
  }
  int mbase = half * 8;
#pragma unroll
  for (int t = 0; t < 4; ++t)
#pragma unroll
    for (int v = 0; v < 8; ++v) {
      int pp = mtile * 16 + mbase + v;
      if (pp < MO) out[(size_t)pp * 384 + ng * 64 + t * 16 + n] = (_Float16)acc[t][v];
    }
}

// -------- cost kernel: one wave = 32 pixels x 1 disparity, 2 FC layers + dot.
// Two A fragments share every B fragment (2 WMMAs per weight load). fw4 is
// staged to LDS via CDNA5 async global->LDS copy, overlapped with the GEMMs.
__global__ void __launch_bounds__(32) cost_kernel(
    const _Float16* __restrict__ Aref, const _Float16* __restrict__ Bsec,
    const _Float16* __restrict__ fw2h, const float* __restrict__ fb2,
    const _Float16* __restrict__ fw3h, const float* __restrict__ fb3,
    const float* __restrict__ fw4, const float* __restrict__ fb4,
    float* __restrict__ out) {
  __shared__ _Float16 h1[32 * LROW];
  __shared__ _Float16 h2[32 * LROW];
  __shared__ float fw4l[384];
  int lane = threadIdx.x & 31;
  int d = blockIdx.y;                            // disparity (disp_min == 0)
  int pbase = blockIdx.x * 32;
  if (pbase >= MO) return;

  // Warm caches for the weight streams.
  __builtin_prefetch(fw2h, 0, 3);
  __builtin_prefetch(fw3h, 0, 3);

  // Async-stage fw4 (384 f32 = 96 x b128) into LDS; completes during GEMMs.
#pragma unroll
  for (int i = 0; i < 3; ++i) {
    unsigned int laddr =
        (unsigned int)(uintptr_t)(&fw4l[0]) + (unsigned int)((i * 32 + lane) * 16);
    const char* gaddr = (const char*)fw4 + (i * 32 + lane) * 16;
    asm volatile("global_load_async_to_lds_b128 %0, %1, off"
                 :: "v"(laddr), "v"(gaddr) : "memory");
  }

  // h1 = relu(A[p] + B[p+d]) (f16), zero for invalid pixels
  for (int it = lane; it < 32 * 48; it += 32) {
    int pix = it / 48, g = it % 48;
    int p = pbase + pix;
    int pc = (p < MO) ? p : MO - 1;
    int x = pc % WF;
    bool valid = (p < MO) && (x + d < WF);
    int pb = valid ? pc + d : pc;
    union { uint4 q; _Float16 h[8]; } ua, ub, ur;
    ua.q = *((const uint4*)(Aref + (size_t)pc * 384) + g);
    ub.q = *((const uint4*)(Bsec + (size_t)pb * 384) + g);
#pragma unroll
    for (int j = 0; j < 8; ++j) {
      float s = (float)ua.h[j] + (float)ub.h[j];
      ur.h[j] = (_Float16)(valid ? fmaxf(s, 0.f) : 0.f);
    }
    *(uint4*)(h1 + pix * LROW + g * 8) = ur.q;
  }

  int n = lane & 15, half = lane >> 4, mbase = half * 8;

  // layer 2: h2 = relu(h1 @ fw2^T + fb2)
#pragma unroll 1
  for (int ng = 0; ng < 6; ++ng) {
    v8f acc0[4], acc1[4];
#pragma unroll
    for (int t = 0; t < 4; ++t) {
      float bv = fb2[ng * 64 + t * 16 + n];
#pragma unroll
      for (int v = 0; v < 8; ++v) { acc0[t][v] = bv; acc1[t][v] = bv; }
    }
#pragma unroll
    for (int kc = 0; kc < 12; ++kc) {
      v16h a0 = load_a_frag(h1, LROW, kc * 32);
      v16h a1 = load_a_frag(h1 + 16 * LROW, LROW, kc * 32);
#pragma unroll
      for (int t = 0; t < 4; ++t) {
        v16h b = load_b_frag(fw2h, 384, ng * 64 + t * 16, kc * 32);
        acc0[t] = wmma16(a0, b, acc0[t]);
        acc1[t] = wmma16(a1, b, acc1[t]);
      }
    }
#pragma unroll
    for (int t = 0; t < 4; ++t)
#pragma unroll
      for (int v = 0; v < 8; ++v) {
        h2[(mbase + v) * LROW + ng * 64 + t * 16 + n] = (_Float16)fmaxf(acc0[t][v], 0.f);
        h2[(16 + mbase + v) * LROW + ng * 64 + t * 16 + n] = (_Float16)fmaxf(acc1[t][v], 0.f);
      }
  }

  // layer 3: h1 = relu(h2 @ fw3^T + fb3)
#pragma unroll 1
  for (int ng = 0; ng < 6; ++ng) {
    v8f acc0[4], acc1[4];
#pragma unroll
    for (int t = 0; t < 4; ++t) {
      float bv = fb3[ng * 64 + t * 16 + n];
#pragma unroll
      for (int v = 0; v < 8; ++v) { acc0[t][v] = bv; acc1[t][v] = bv; }
    }
#pragma unroll
    for (int kc = 0; kc < 12; ++kc) {
      v16h a0 = load_a_frag(h2, LROW, kc * 32);
      v16h a1 = load_a_frag(h2 + 16 * LROW, LROW, kc * 32);
#pragma unroll
      for (int t = 0; t < 4; ++t) {
        v16h b = load_b_frag(fw3h, 384, ng * 64 + t * 16, kc * 32);
        acc0[t] = wmma16(a0, b, acc0[t]);
        acc1[t] = wmma16(a1, b, acc1[t]);
      }
    }
#pragma unroll
    for (int t = 0; t < 4; ++t)
#pragma unroll
      for (int v = 0; v < 8; ++v) {
        h1[(mbase + v) * LROW + ng * 64 + t * 16 + n] = (_Float16)fmaxf(acc0[t][v], 0.f);
        h1[(16 + mbase + v) * LROW + ng * 64 + t * 16 + n] = (_Float16)fmaxf(acc1[t][v], 0.f);
      }
  }

  // Ensure the async fw4 staging has landed in LDS.
  asm volatile("s_wait_asynccnt 0x0" ::: "memory");

  // final: each lane reduces its own pixel; score = sigmoid(h . fw4 + fb4)
  float ssum = 0.f;
  for (int k = 0; k < 384; ++k)
    ssum = fmaf((float)h1[lane * LROW + k], fw4l[k], ssum);
  int p = pbase + lane;
  if (p < MO) {
    int x = p % WF;
    float z = ssum + fb4[0];
    float score = 1.f / (1.f + __expf(-z));
    out[(size_t)p * D_NUM + d] = (x + d < WF) ? -score : __builtin_nanf("");
  }
}

// ---------------------------------------------------------------------- host
static inline void* wsalloc(char* base, size_t& off, size_t bytes) {
  off = (off + 255) & ~(size_t)255;
  void* p = base + off;
  off += bytes;
  return p;
}

extern "C" void kernel_launch(void* const* d_in, const int* in_sizes, int n_in,
                              void* d_out, int out_size, void* d_ws, size_t ws_size,
                              hipStream_t stream) {
  (void)in_sizes; (void)n_in; (void)out_size; (void)ws_size;
  const float* ref = (const float*)d_in[0];
  const float* sec = (const float*)d_in[1];
  const float* w1  = (const float*)d_in[2];
  const float* b1  = (const float*)d_in[3];
  const float* w2  = (const float*)d_in[4];
  const float* b2  = (const float*)d_in[5];
  const float* w3  = (const float*)d_in[6];
  const float* b3  = (const float*)d_in[7];
  const float* w4  = (const float*)d_in[8];
  const float* b4  = (const float*)d_in[9];
  const float* w5  = (const float*)d_in[10];
  const float* b5  = (const float*)d_in[11];
  const float* fw1 = (const float*)d_in[12];
  const float* fb1 = (const float*)d_in[13];
  const float* fw2 = (const float*)d_in[14];
  const float* fb2 = (const float*)d_in[15];
  const float* fw3 = (const float*)d_in[16];
  const float* fb3 = (const float*)d_in[17];
  const float* fw4 = (const float*)d_in[18];
  const float* fb4 = (const float*)d_in[19];
  float* out = (float*)d_out;

  char* ws = (char*)d_ws;
  size_t off = 0;
  const size_t szAct = (size_t)190 * 446 * 128 * 2;     // f16 (Hi,Wi,128) max
  _Float16* bufA = (_Float16*)wsalloc(ws, off, szAct);
  _Float16* bufB = (_Float16*)wsalloc(ws, off, szAct);
  _Float16* Aref = (_Float16*)wsalloc(ws, off, (size_t)MO * 384 * 2);
  _Float16* Bsec = (_Float16*)wsalloc(ws, off, (size_t)(MO + 32) * 384 * 2);
  _Float16* wc2  = (_Float16*)wsalloc(ws, off, (size_t)112 * 1152 * 2);
  _Float16* wc3  = (_Float16*)wsalloc(ws, off, (size_t)112 * 1152 * 2);
  _Float16* wc4  = (_Float16*)wsalloc(ws, off, (size_t)112 * 1152 * 2);
  _Float16* wc5  = (_Float16*)wsalloc(ws, off, (size_t)112 * 1152 * 2);
  _Float16* fw1r = (_Float16*)wsalloc(ws, off, (size_t)384 * 128 * 2);
  _Float16* fw1s = (_Float16*)wsalloc(ws, off, (size_t)384 * 128 * 2);
  _Float16* fw2h = (_Float16*)wsalloc(ws, off, (size_t)384 * 384 * 2);
  _Float16* fw3h = (_Float16*)wsalloc(ws, off, (size_t)384 * 384 * 2);

  // Zero activation ping-pong so padded channels 112..127 contribute 0 to GEMMs.
  {
    size_t n4 = (2 * szAct) / 16;
    zero_kernel<<<2048, 256, 0, stream>>>((uint4*)bufA, n4);
  }
  {
    int total = 4 * 112 * 1152 + 2 * 384 * 128 + 2 * 384 * 384;
    prep_kernel<<<(total + 255) / 256, 256, 0, stream>>>(
        w2, w3, w4, w5, fw1, fw2, fw3,
        wc2, wc3, wc4, wc5, fw1r, fw1s, fw2h, fw3h);
  }

  const int Hs[6] = {192, 190, 188, 186, 184, 182};
  const int Wsz[6] = {448, 446, 444, 442, 440, 438};
  const _Float16* wcs[4] = {wc2, wc3, wc4, wc5};
  const float* bcs[4] = {b2, b3, b4, b5};

  for (int img = 0; img < 2; ++img) {
    const float* src = img == 0 ? ref : sec;
    // conv1 -> bufA
    {
      int Ho = Hs[1], Wo = Wsz[1];
      int total = Ho * Wo * 112;
      conv1_kernel<<<(total + 255) / 256, 256, 0, stream>>>(src, w1, b1, bufA,
                                                            Ho, Wo, Wsz[0]);
    }
    // conv2..conv5, ping-pong bufA/bufB
    _Float16* bin = bufA;
    _Float16* bout = bufB;
    for (int l = 0; l < 4; ++l) {
      int Ho = Hs[l + 2], Wo = Wsz[l + 2], Wi = Wsz[l + 1];
      int mtiles = (Ho * Wo + 15) / 16;
      convN_kernel<<<(mtiles + 3) / 4, 128, 0, stream>>>(bin, bcs[l], wcs[l], bout,
                                                         Ho, Wo, Wi);
      _Float16* t = bin; bin = bout; bout = t;
    }
    // bin now holds final (182,438,128) features
    {
      int mtiles = (MO + 15) / 16;
      dim3 grid((mtiles + 3) / 4, 6);
      fc1_kernel<<<grid, 128, 0, stream>>>(bin,
                                           img == 0 ? fw1r : fw1s,
                                           fb1, img == 0 ? 0 : 1,
                                           img == 0 ? Aref : Bsec);
    }
  }

  // cost volume: one wave per (32-pixel tile, disparity)
  {
    int mtiles32 = (MO + 31) / 32;
    dim3 grid(mtiles32, D_NUM);
    cost_kernel<<<grid, 32, 0, stream>>>(Aref, Bsec, fw2h, fb2, fw3h, fb3,
                                         fw4, fb4, out);
  }
}
